// RWKVBackbone_644245094450
// MI455X (gfx1250) — compile-verified, matrix operand order checked by reading, not run on hardware
//
#include <hip/hip_runtime.h>
#include <hip/hip_bf16.h>
#include <math.h>

// ---------------------------------------------------------------------------
// LN + 2-layer GRU (PyTorch gate order r,z,n) for B=32, T=2048, H=512.
// bf16 WMMA (v_wmma_f32_16x16x32_bf16) for all matmuls, f32 accumulate.
// All intermediates are TIME-MAJOR:
//   hA/hB (bf16 GEMM inputs):  row rt = t*B + b, [rt][H]
//   xp (fp32 projections):     [t][gate][u][b]  -> coalesced scan reads and
//                              contiguous GEMM epilogue stores.
// ---------------------------------------------------------------------------

typedef __attribute__((ext_vector_type(16))) __bf16 v16bf;
typedef __attribute__((ext_vector_type(8)))  __bf16 v8bf;
typedef __attribute__((ext_vector_type(8)))  float  v8f;

#define HH   512
#define BB   32
#define TSTEPS 2048
#define RR   (BB * TSTEPS)     // 65536 rows
#define G3   (3 * HH)          // 1536
#define XPSTEP (G3 * BB)       // floats per timestep block of xp
#define NWG  32                // workgroups in persistent scan
#define NU   16                // hidden units per workgroup (NWG*NU == HH)

// Build a 16x-bf16 WMMA operand from two contiguous 16-byte chunks at
// k and k+16 (per CDNA5 16-bit A/B VGPR layout).
static __device__ inline v16bf ld_bf16x16(const __hip_bfloat16* p) {
  v8bf lo = *(const v8bf*)(const void*)(p);
  v8bf hi = *(const v8bf*)(const void*)(p + 16);
  return __builtin_shufflevector(lo, hi, 0, 1, 2, 3, 4, 5, 6, 7,
                                 8, 9, 10, 11, 12, 13, 14, 15);
}

// ---------------- weight fp32 -> bf16 --------------------------------------
__global__ void cvt_w_kernel(const float* __restrict__ wih,
                             const float* __restrict__ whh,
                             __hip_bfloat16* __restrict__ wih_bf,
                             __hip_bfloat16* __restrict__ whh_bf, int n) {
  int i = blockIdx.x * blockDim.x + threadIdx.x;
  if (i < n) {
    wih_bf[i] = __float2bfloat16(wih[i]);
    whh_bf[i] = __float2bfloat16(whh[i]);
  }
}

// ---------------- LayerNorm: one wave32 per row of 512 ---------------------
// Reads x in [b][t][H]; writes bf16 rows time-major: rt = t*B + b.
__global__ __launch_bounds__(256) void ln_kernel(const float* __restrict__ x,
                                                 const float* __restrict__ gamma,
                                                 const float* __restrict__ beta,
                                                 __hip_bfloat16* __restrict__ out) {
  const int wave = threadIdx.x >> 5, lane = threadIdx.x & 31;
  const int r = blockIdx.x * 8 + wave;         // r = b*T + t
  const int b = r >> 11;                       // / TSTEPS
  const int t = r & (TSTEPS - 1);
  const float* xr = x + (size_t)r * HH;
  float v[16], s = 0.f, ss = 0.f;
#pragma unroll
  for (int i = 0; i < 16; ++i) {
    float q = xr[lane + 32 * i];
    v[i] = q; s += q; ss += q * q;
  }
#pragma unroll
  for (int m = 16; m >= 1; m >>= 1) {
    s  += __shfl_xor(s, m, 32);
    ss += __shfl_xor(ss, m, 32);
  }
  const float mu  = s * (1.f / HH);
  const float var = ss * (1.f / HH) - mu * mu;
  const float inv = rsqrtf(var + 1e-5f);
  __hip_bfloat16* orow = out + (size_t)(t * BB + b) * HH;
#pragma unroll
  for (int i = 0; i < 16; ++i) {
    int c = lane + 32 * i;
    orow[c] = __float2bfloat16((v[i] - mu) * inv * gamma[c] + beta[c]);
  }
}

// ---------------- xp = A[rt,H] @ W^T[3H,H] + b : WMMA GEMM -----------------
// Block = 64 rows x 64 cols; 8 waves, each owns a 16-row strip and two
// 16-wide N tiles.  Output goes to xp[t][gate][u][b]: since rt = t*B + b,
// consecutive accumulator rows are consecutive b -> contiguous stores.
__global__ __launch_bounds__(256) void gemm_xp_kernel(
    const __hip_bfloat16* __restrict__ A,     // [RR, HH] time-major rows
    const __hip_bfloat16* __restrict__ Wbf,   // [G3, HH]
    const float* __restrict__ bias,           // [G3]
    float* __restrict__ xp) {                 // [T][3][HH][BB]
  const int w = threadIdx.x >> 5, lane = threadIdx.x & 31;
  const int ln16 = lane & 15, kh = lane >> 4;       // kh selects K half
  const int rowbase = blockIdx.x * 64 + (w & 3) * 16;
  const int col0 = blockIdx.y * 64 + (w >> 2) * 32; // two tiles: col0, col0+16

  v8f acc0 = {}; v8f acc1 = {};
  const __hip_bfloat16* Arow = A   + (size_t)(rowbase + ln16) * HH;
  const __hip_bfloat16* B0   = Wbf + (size_t)(col0 + ln16) * HH;
  const __hip_bfloat16* B1   = Wbf + (size_t)(col0 + 16 + ln16) * HH;

  for (int k0 = 0; k0 < HH; k0 += 32) {
    const int koff = k0 + (kh << 3);
    v16bf a  = ld_bf16x16(Arow + koff);
    v16bf b0 = ld_bf16x16(B0 + koff);
    v16bf b1 = ld_bf16x16(B1 + koff);
    acc0 = __builtin_amdgcn_wmma_f32_16x16x32_bf16(false, a, false, b0,
                                                   (short)0, acc0, false, false);
    acc1 = __builtin_amdgcn_wmma_f32_16x16x32_bf16(false, a, false, b1,
                                                   (short)0, acc1, false, false);
  }
  const int c0 = col0 + ln16, c1 = col0 + 16 + ln16;   // cols in [0, 1536)
  const float bi0 = bias[c0], bi1 = bias[c1];
  const int rt0 = rowbase + 8 * kh;                    // rows rt0..rt0+7
  const int t = rt0 >> 5;                              // rt / 32
  const int b0i = rt0 & 31;                            // stays in [0,32) for v<8
  float* dst0 = xp + (size_t)t * XPSTEP + (size_t)c0 * BB + b0i;
  float* dst1 = xp + (size_t)t * XPSTEP + (size_t)c1 * BB + b0i;
#pragma unroll
  for (int v = 0; v < 8; ++v) {
    dst0[v] = acc0[v] + bi0;
    dst1[v] = acc1[v] + bi1;
  }
}

// ---------------- device-scope grid barrier (monotonic counter) ------------
static __device__ inline void grid_sync(unsigned* bar, unsigned target) {
  __syncthreads();
  if (threadIdx.x == 0) {
    __hip_atomic_fetch_add(bar, 1u, __ATOMIC_RELEASE, __HIP_MEMORY_SCOPE_AGENT);
    while (__hip_atomic_load(bar, __ATOMIC_ACQUIRE, __HIP_MEMORY_SCOPE_AGENT) <
           target) {
      __builtin_amdgcn_s_sleep(2);
    }
  }
  __syncthreads();
}

// ---------------- persistent GRU scan --------------------------------------
// NWG workgroups, 6 waves each.  WG g owns units [g*NU, g*NU+NU); its 48 W_hh
// rows (r,z,n) live in LDS for all T steps.  Per step: wave w computes the
// (gate=w>>1, mtile=w&1) 16x16 hp tile over K=512 with 16 WMMAs, stashes it in
// LDS, then fused gate math (coalesced xp reads, lane==batch) writes h_new;
// double-buffered h state + grid sync.
__global__ __launch_bounds__(192) void gru_scan_kernel(
    const float* __restrict__ xp,            // [T][3][HH][BB] this layer
    const __hip_bfloat16* __restrict__ Whh,  // [G3, HH] bf16 this layer
    const float* __restrict__ bhh,           // [G3]
    __hip_bfloat16* __restrict__ hseq,       // [T*B rows][HH] bf16 (next layer A)
    float* __restrict__ outf,                // fp32 out [B][T][HH] (last layer)
    __hip_bfloat16* __restrict__ hbuf,       // [2][BB][HH] bf16 state
    unsigned* __restrict__ bar, int write_f32) {
  __shared__ __hip_bfloat16 Wlds[48 * HH];   // 48 KB
  __shared__ float hpL[48 * BB];             // 6 KB: [gate*16+ul][batch]

  const int tid = threadIdx.x;
  const int u0 = blockIdx.x * NU;

  // Stage W_hh slice: local row lr -> global row (lr/16)*H + u0 + (lr%16).
  for (int c = tid; c < 48 * (HH / 8); c += 192) {
    const int lr = c >> 6, cc = c & 63;
    const int grow = (lr >> 4) * HH + u0 + (lr & 15);
    *(v8bf*)(void*)&Wlds[lr * HH + cc * 8] =
        *(const v8bf*)(const void*)&Whh[(size_t)grow * HH + cc * 8];
  }
  // h0 = 0 for our unit slice, all batches, buffer 0.
  for (int i = tid; i < NU * BB; i += 192) {
    const int ul = i >> 5, b = i & 31;
    hbuf[(size_t)b * HH + u0 + ul] = __float2bfloat16(0.f);
  }
  __threadfence();
  grid_sync(bar, NWG);  // arrival 1: h0 visible everywhere

  const int w = tid >> 5, lane = tid & 31;
  const int ln16 = lane & 15, kh = lane >> 4;
  const int g = w >> 1, mt = w & 1;  // gate tile, batch tile
  const __hip_bfloat16* Brow = &Wlds[(g * 16 + ln16) * HH];

  for (int t = 0; t < TSTEPS; ++t) {
    const __hip_bfloat16* hrd = hbuf + (size_t)(t & 1) * (BB * HH);
    __hip_bfloat16* hwr = hbuf + (size_t)((t + 1) & 1) * (BB * HH);

    // hp tile: M=batch, N=unit, K=512
    v8f acc = {};
    const __hip_bfloat16* Arow = hrd + (size_t)(mt * 16 + ln16) * HH;
#pragma unroll 4
    for (int k0 = 0; k0 < HH; k0 += 32) {
      const int koff = k0 + (kh << 3);
      v16bf a = ld_bf16x16(Arow + koff);
      v16bf b = ld_bf16x16(Brow + koff);
      acc = __builtin_amdgcn_wmma_f32_16x16x32_bf16(false, a, false, b,
                                                    (short)0, acc, false, false);
    }
    {
      float* dst = &hpL[(g * 16 + ln16) * BB + mt * 16 + 8 * kh];
#pragma unroll
      for (int v = 0; v < 8; ++v) dst[v] = acc[v];
    }
    __syncthreads();

    // fused gates for our 16 units x 32 batches; xp reads coalesced in b.
    const float* xbase = xp + (size_t)t * XPSTEP;
    for (int i = tid; i < NU * BB; i += 192) {
      const int ul = i >> 5, b = i & 31;
      const int u = u0 + ul;
      const float xr = xbase[((size_t)u) * BB + b];
      const float xz = xbase[((size_t)(HH + u)) * BB + b];
      const float xn = xbase[((size_t)(2 * HH + u)) * BB + b];
      const float hpr = hpL[(0 * 16 + ul) * BB + b] + bhh[u];
      const float hpz = hpL[(16 + ul) * BB + b] + bhh[HH + u];
      const float hpn = hpL[(32 + ul) * BB + b] + bhh[2 * HH + u];
      const float hprev = __bfloat162float(hrd[(size_t)b * HH + u]);
      const float rg = 1.f / (1.f + __expf(-(xr + hpr)));
      const float zg = 1.f / (1.f + __expf(-(xz + hpz)));
      const float ng = tanhf(xn + rg * hpn);
      const float hn = (1.f - zg) * ng + zg * hprev;
      hwr[(size_t)b * HH + u] = __float2bfloat16(hn);
      hseq[((size_t)t * BB + b) * HH + u] = __float2bfloat16(hn);
      if (write_f32) outf[((size_t)b * TSTEPS + t) * HH + u] = hn;
      if (t + 1 < TSTEPS) {
        __builtin_prefetch(xbase + XPSTEP + (size_t)u * BB + b, 0, 1);
        __builtin_prefetch(xbase + XPSTEP + (size_t)(2 * HH + u) * BB + b, 0, 1);
      }
    }
    __threadfence();
    grid_sync(bar, (unsigned)NWG * (t + 2));
  }
}

// ---------------------------------------------------------------------------
extern "C" void kernel_launch(void* const* d_in, const int* in_sizes, int n_in,
                              void* d_out, int out_size, void* d_ws,
                              size_t ws_size, hipStream_t stream) {
  (void)in_sizes; (void)n_in; (void)out_size; (void)ws_size;
  const float* x     = (const float*)d_in[0];
  const float* gamma = (const float*)d_in[1];
  const float* beta  = (const float*)d_in[2];
  const float* W_ih  = (const float*)d_in[3];  // [2, 1536, 512]
  const float* W_hh  = (const float*)d_in[4];  // [2, 1536, 512]
  const float* b_ih  = (const float*)d_in[5];  // [2, 1536]
  const float* b_hh  = (const float*)d_in[6];  // [2, 1536]
  float* out = (float*)d_out;                  // [B, T, H]

  char* ws = (char*)d_ws;
  size_t off = 0;
  unsigned* bar = (unsigned*)(ws + off);                 off += 256;
  __hip_bfloat16* hbuf   = (__hip_bfloat16*)(ws + off);  off += (size_t)2 * BB * HH * 2;
  __hip_bfloat16* wih_bf = (__hip_bfloat16*)(ws + off);  off += (size_t)2 * G3 * HH * 2;
  __hip_bfloat16* whh_bf = (__hip_bfloat16*)(ws + off);  off += (size_t)2 * G3 * HH * 2;
  __hip_bfloat16* hA     = (__hip_bfloat16*)(ws + off);  off += (size_t)RR * HH * 2;
  __hip_bfloat16* hB     = (__hip_bfloat16*)(ws + off);  off += (size_t)RR * HH * 2;
  float*          xp     = (float*)(ws + off);           off += (size_t)RR * G3 * 4;

  hipMemsetAsync(bar, 0, 256, stream);  // reset grid-barrier counters

  const int nW = 2 * G3 * HH;
  cvt_w_kernel<<<(nW + 255) / 256, 256, 0, stream>>>(W_ih, W_hh, wih_bf,
                                                     whh_bf, nW);
  ln_kernel<<<RR / 8, 256, 0, stream>>>(x, gamma, beta, hA);

  dim3 gg(RR / 64, G3 / 64);
  // layer 0
  gemm_xp_kernel<<<gg, 256, 0, stream>>>(hA, wih_bf, b_ih, xp);
  gru_scan_kernel<<<NWG, 192, 0, stream>>>(xp, whh_bf, b_hh, hB, out, hbuf,
                                           bar + 0, 0);
  // layer 1
  gemm_xp_kernel<<<gg, 256, 0, stream>>>(hB, wih_bf + (size_t)G3 * HH,
                                         b_ih + G3, xp);
  gru_scan_kernel<<<NWG, 192, 0, stream>>>(xp, whh_bf + (size_t)G3 * HH,
                                           b_hh + G3, hA, out, hbuf, bar + 1, 1);
}
